// BNNFC_32856499814961
// MI455X (gfx1250) — matrix operand order repaired and checked by reading, compiled, not run on hardware
//
#include <hip/hip_runtime.h>
#include <cstdint>

// ---------------------------------------------------------------------------
// GLIFR (BNNFC) on gfx1250.
// Exploits the 20-step synaptic delay: 50 chunks of 20 timesteps.
// Per chunk:
//   kernel A (fused): syn[640x512] = [fire_hist | x] (640x768 bf16)
//                     @ WcatT (768x512 bf16)  via v_wmma_f32_16x16x32_bf16,
//                     then 20-step elementwise GLIFR recurrence in LDS/regs,
//                     writing bf16 firing into a 40-slot ring.
//   kernel B: out[640x128] = firing (640x512) @ out_w (512x128) + bias (WMMA).
// Weight slice staging into LDS now uses the Tensor Data Mover
// (tensor_load_to_lds + s_wait_tensorcnt) when the builtin is available.
// ---------------------------------------------------------------------------

#define DT_C   0.05f
#define R_C    0.1f

typedef __attribute__((ext_vector_type(16))) __bf16 bf16x16;
typedef __attribute__((ext_vector_type(8)))  float  f32x8;
typedef __attribute__((ext_vector_type(4)))  unsigned int u32x4;
typedef __attribute__((ext_vector_type(8)))  int    i32x8;
typedef __attribute__((ext_vector_type(4)))  int    i32x4;

static __device__ __forceinline__ uint16_t f2bf(float x) {
  uint32_t u = __float_as_uint(x);
  uint32_t r = u + 0x7FFFu + ((u >> 16) & 1u);   // round-to-nearest-even
  return (uint16_t)(r >> 16);
}
static __device__ __forceinline__ float sigmoidf(float x) {
  return 1.0f / (1.0f + __expf(-x));
}

// A fragment (16-bit A 16x32 layout): lanes 0-15 hold K {0..7,16..23},
// lanes 16-31 hold K {8..15,24..31} of their row. Two 16B loads.
static __device__ __forceinline__ bf16x16 load_a_frag(const uint16_t* row, int kBase, int hi8) {
  bf16x16 f;
  ((uint4*)&f)[0] = *(const uint4*)(row + kBase + hi8);
  ((uint4*)&f)[1] = *(const uint4*)(row + kBase + 16 + hi8);
  return f;
}
// B fragment (16-bit B 32x16): lanes 0-15 hold K {0..7,16..23}, lanes 16-31
// K {8..15,24..31} of their column (same as A). Two 16B loads.
static __device__ __forceinline__ bf16x16 load_b_frag(const uint16_t* col, int base) {
  bf16x16 f;
  ((uint4*)&f)[0] = *(const uint4*)(col + base);
  ((uint4*)&f)[1] = *(const uint4*)(col + base + 16);
  return f;
}

// ---------------- prep kernels --------------------------------------------

// WcatT[h][k] (512x768): k<512 -> W_lat[k][h], k>=512 -> W_iv[k-512][h]
// outwT[o][h] (128x512): out_w[h][o]
__global__ void glifr_prep_weights(const float* __restrict__ wlat,
                                   const float* __restrict__ wiv,
                                   const float* __restrict__ outw,
                                   uint16_t* __restrict__ WcatT,
                                   uint16_t* __restrict__ outwT) {
  int idx = blockIdx.x * 256 + threadIdx.x;
  if (idx < 512 * 768) {
    int h = idx / 768, k = idx % 768;
    float v = (k < 512) ? wlat[k * 512 + h] : wiv[(k - 512) * 512 + h];
    WcatT[idx] = f2bf(v);
  } else {
    int j = idx - 512 * 768;           // < 128*512 by grid sizing
    int o = j / 512, h = j % 512;
    outwT[j] = f2bf(outw[h * 128 + o]);
  }
}

// xbf[t][b][k] = bf16(input[b][t][k]); 4 elems per thread
__global__ void glifr_prep_input(const float* __restrict__ inp,
                                 uint16_t* __restrict__ xbf) {
  int i = blockIdx.x * 256 + threadIdx.x;   // 2,048,000 quads
  int s = i * 4;
  int b   = s / 256000;
  int rem = s % 256000;
  int t = rem >> 8;
  int k = rem & 255;
  float4 v = *(const float4*)(inp + s);
  size_t d = ((size_t)(t * 32 + b)) * 256 + k;
  xbf[d + 0] = f2bf(v.x); xbf[d + 1] = f2bf(v.y);
  xbf[d + 2] = f2bf(v.z); xbf[d + 3] = f2bf(v.w);
}

__global__ void glifr_zero(uint32_t* __restrict__ p, int n) {
  int i = blockIdx.x * 256 + threadIdx.x;
  if (i < n) p[i] = 0u;
}

// ---------------- fused chunk kernel: GEMM + 20-step recurrence ------------
// grid = 16 WGs (32 columns each), block = 256 threads (8 waves).
// LDS: synLDS 640*32 f32 (80KB) + Btile 32*768 bf16 (48KB) = 128KB.
#define BTILE_LDS_OFF (640 * 32 * 4)   // 81920: Btile byte offset in LDS

__launch_bounds__(256, 1)
__global__ void glifr_chunk_kernel(const uint16_t* __restrict__ WcatT,
                                   const uint16_t* __restrict__ xbf,
                                   uint16_t* __restrict__ hist,    // [40][32][512] bf16 ring
                                   float* __restrict__ volt_s,     // [32][512]
                                   float* __restrict__ fire_s,     // [32][512]
                                   float* __restrict__ asc_s,      // [2][32][512]
                                   const float* __restrict__ thresh,
                                   const float* __restrict__ tkm,
                                   const float* __restrict__ ascamp,
                                   const float* __restrict__ tascr,
                                   const float* __restrict__ tkasc,
                                   int t0) {
  extern __shared__ char smem[];
  float*    synLDS = (float*)smem;                         // 640*32 f32
  uint16_t* Btile  = (uint16_t*)(smem + BTILE_LDS_OFF);    // 32*768 bf16

  const int tid = threadIdx.x;
  const int h0  = blockIdx.x * 32;
  const int lane = tid & 31, wave = tid >> 5;

  // ---- Stage this WG's 32-column weight slice (contiguous rows of WcatT)
  //      into LDS. Preferred path: Tensor Data Mover (async DMA, TENSORcnt).
#if __has_builtin(__builtin_amdgcn_tensor_load_to_lds)
  if (wave == 0) {
    const uint64_t ga = (uint64_t)(uintptr_t)(WcatT + (size_t)h0 * 768);
    // D# group 0: count=1 (valid), lds_addr, global_addr[56:0], type=2
    u32x4 g0 = { 1u,
                 (unsigned int)BTILE_LDS_OFF,
                 (unsigned int)(ga & 0xFFFFFFFFu),
                 (unsigned int)((ga >> 32) & 0x01FFFFFFu) | 0x80000000u };
    // D# group 1: data_size=2B; tensor 768x512; tile 768x32; dim0 stride 768
    i32x8 g1 = { (int)0x00010000u,        // workgroup_mask=0, data_size=1 (2B)
                 (int)(768u << 16),       // tensor_dim0[15:0] in [31:16]
                 (int)(512u << 16),       // tensor_dim0 hi=0, tensor_dim1 lo
                 (int)(768u << 16),       // tensor_dim1 hi=0, tile_dim0=768
                 32,                      // tile_dim1=32, tile_dim2=0
                 768,                     // tensor_dim0_stride lo32
                 0, 0 };                  // stride hi, tensor_dim1_stride=0
    i32x4 gz = { 0, 0, 0, 0 };            // 2-D tensor: groups 2/3 unused
#if __clang_major__ >= 23
    i32x8 gz8 = { 0, 0, 0, 0, 0, 0, 0, 0 };
    __builtin_amdgcn_tensor_load_to_lds(g0, g1, gz, gz, gz8, 0);
#else
    __builtin_amdgcn_tensor_load_to_lds(g0, g1, gz, gz, 0);
#endif
    __builtin_amdgcn_s_wait_tensorcnt(0);
  }
#else
  {
    const uint4* src = (const uint4*)(WcatT + (size_t)h0 * 768);
    uint4*       dst = (uint4*)Btile;
#pragma unroll
    for (int i = 0; i < 12; ++i) dst[tid + 256 * i] = src[tid + 256 * i];
  }
#endif
  __syncthreads();

  const int laneLo = lane & 15, hiHalf = lane >> 4;
  const int hi8 = hiHalf * 8, hi16 = hiHalf * 16;

  // ---- Phase A: syn = Act(640x768) @ Wslice(768x32), K-concat of lateral+ff
  for (int rt = wave; rt < 40; rt += 8) {           // 40 row-tiles of 16
    const int t_local = rt >> 1;
    const int b       = ((rt & 1) << 4) + laneLo;   // row = t_local*32 + b
    const int slotR   = (t0 + t_local) % 40;        // firing[t-20]
    const uint16_t* histRow = hist + ((size_t)(slotR * 32 + b)) * 512;
    const uint16_t* xRow    = xbf  + ((size_t)((t0 + t_local) * 32 + b)) * 256;
    __builtin_prefetch(histRow, 0, 1);
    __builtin_prefetch(xRow, 0, 1);
    const uint16_t* colA = Btile + (size_t)laneLo * 768;          // ct=0
    const uint16_t* colB = Btile + (size_t)(16 + laneLo) * 768;   // ct=1
    f32x8 acc0 = {};
    f32x8 acc1 = {};
#pragma unroll 4
    for (int ks = 0; ks < 24; ++ks) {               // K = 768 = 24 * 32
      const uint16_t* aRow = (ks < 16) ? histRow : xRow;
      const int kBase      = (ks < 16) ? ks * 32 : (ks - 16) * 32;
      bf16x16 a  = load_a_frag(aRow, kBase, hi8);
      bf16x16 b0 = load_b_frag(colA, ks * 32 + hi8);
      bf16x16 b1 = load_b_frag(colB, ks * 32 + hi8);
      acc0 = __builtin_amdgcn_wmma_f32_16x16x32_bf16(false, a, false, b0,
                                                     (short)0, acc0, false, false);
      acc1 = __builtin_amdgcn_wmma_f32_16x16x32_bf16(false, a, false, b1,
                                                     (short)0, acc1, false, false);
    }
#pragma unroll
    for (int v = 0; v < 8; ++v) {                   // C layout: M=v+hi8, N=laneLo
      int r = rt * 16 + v + hi8;
      synLDS[r * 32 + laneLo]      = acc0[v];
      synLDS[r * 32 + 16 + laneLo] = acc1[v];
    }
  }
  __syncthreads();

  // ---- Phase B: 20-step GLIFR recurrence, elementwise in (b, h)
  const int hl = tid & 31;
  const int h  = h0 + hl;
  const int bS = tid >> 5;                          // 0..7
  const float thr  = thresh[h];
  const float km_s = sigmoidf(tkm[h]);              // = DT*k_m
  const float a_m  = 1.0f - km_s;
  const float b_m  = km_s * R_C;                    // DT*k_m*R
  float amp0 = ascamp[h],       amp1 = ascamp[512 + h];
  float ar0  = 1.0f - 2.0f * sigmoidf(tascr[h]);
  float ar1  = 1.0f - 2.0f * sigmoidf(tascr[512 + h]);
  float kd0  = 1.0f - sigmoidf(tkasc[h]);           // 1 - DT*k_asc
  float kd1  = 1.0f - sigmoidf(tkasc[512 + h]);

#pragma unroll
  for (int i = 0; i < 4; ++i) {
    const int b    = bS + 8 * i;
    const int sIdx = b * 512 + h;
    float volt = volt_s[sIdx];
    float fire = fire_s[sIdx];
    float a0   = asc_s[sIdx];
    float a1   = asc_s[16384 + sIdx];
    for (int tl = 0; tl < 20; ++tl) {
      float syn = synLDS[(tl * 32 + b) * 32 + hl];
      a0 = (ar0 * a0 + amp0) * fire * DT_C + a0 * kd0;
      a1 = (ar1 * a1 + amp1) * fire * DT_C + a1 * kd1;
      volt = (a_m - fire) * volt + b_m * (syn + a0 + a1);  // V_RESET=0, I0=0
      fire = sigmoidf(volt - thr);
      int slotW = (t0 + tl + 20) % 40;
      hist[((size_t)(slotW * 32 + b)) * 512 + h] = f2bf(fire);
    }
    volt_s[sIdx] = volt;
    fire_s[sIdx] = fire;
    asc_s[sIdx]  = a0;
    asc_s[16384 + sIdx] = a1;
  }
}

// ---------------- output projection: out = firing @ out_w + b (WMMA) -------
// grid = 40 row-tiles, block = 128 threads (4 waves x 2 col-tiles = 128 cols)
__launch_bounds__(128, 4)
__global__ void glifr_out_kernel(const uint16_t* __restrict__ hist,
                                 const uint16_t* __restrict__ outwT,
                                 const float* __restrict__ out_b,
                                 float* __restrict__ out, int t0) {
  const int tid = threadIdx.x;
  const int lane = tid & 31, wave = tid >> 5;
  const int laneLo = lane & 15, hiHalf = lane >> 4;
  const int hi8 = hiHalf * 8, hi16 = hiHalf * 16;
  const int rt      = blockIdx.x;
  const int t_local = rt >> 1;
  const int bBase   = (rt & 1) << 4;
  const int b       = bBase + laneLo;
  const int slot    = (t0 + t_local + 20) % 40;   // this chunk's firing
  const uint16_t* histRow = hist + ((size_t)(slot * 32 + b)) * 512;
  const int t = t0 + t_local;

#pragma unroll
  for (int cp = 0; cp < 2; ++cp) {
    const int ct = wave * 2 + cp;
    const int o  = ct * 16 + laneLo;
    const uint16_t* col = outwT + (size_t)o * 512;
    f32x8 acc = {};
#pragma unroll 4
    for (int ks = 0; ks < 16; ++ks) {             // K = 512
      bf16x16 a  = load_a_frag(histRow, ks * 32, hi8);
      bf16x16 bb = load_b_frag(col, ks * 32 + hi8);
      acc = __builtin_amdgcn_wmma_f32_16x16x32_bf16(false, a, false, bb,
                                                    (short)0, acc, false, false);
    }
    const float bias = out_b[o];
#pragma unroll
    for (int v = 0; v < 8; ++v) {
      int brow = bBase + v + hi8;                 // C layout: M=v+hi8
      out[((size_t)brow * 1000 + t) * 128 + o] = acc[v] + bias;
    }
  }
}

// ---------------------------------------------------------------------------

extern "C" void kernel_launch(void* const* d_in, const int* in_sizes, int n_in,
                              void* d_out, int out_size, void* d_ws, size_t ws_size,
                              hipStream_t stream) {
  const float* input = (const float*)d_in[0];
  const float* w_iv  = (const float*)d_in[1];
  const float* w_lat = (const float*)d_in[2];
  const float* thr   = (const float*)d_in[3];
  const float* tkm   = (const float*)d_in[4];
  const float* aamp  = (const float*)d_in[5];
  const float* tar   = (const float*)d_in[6];
  const float* tka   = (const float*)d_in[7];
  const float* outw  = (const float*)d_in[8];
  const float* outb  = (const float*)d_in[9];
  float* out = (float*)d_out;

  // workspace layout (bytes):
  char* ws = (char*)d_ws;
  uint16_t* WcatT  = (uint16_t*)(ws);                 // 512*768*2   =   786,432
  uint16_t* outwT  = (uint16_t*)(ws + 786432);        // 128*512*2   =   131,072
  uint16_t* xbf    = (uint16_t*)(ws + 917504);        // 1000*32*256*2 = 16,384,000
  float*    volt_s = (float*)(ws + 17301504);         // 32*512*4    =    65,536
  float*    fire_s = (float*)(ws + 17367040);         // 32*512*4    =    65,536
  float*    asc_s  = (float*)(ws + 17432576);         // 2*32*512*4  =   131,072
  uint16_t* hist   = (uint16_t*)(ws + 17563648);      // 40*32*512*2 = 1,310,720
  // total: 18,874,368 bytes

  (void)hipFuncSetAttribute((const void*)glifr_chunk_kernel,
                            hipFuncAttributeMaxDynamicSharedMemorySize, 131072);

  glifr_prep_weights<<<1792, 256, 0, stream>>>(w_lat, w_iv, outw, WcatT, outwT);
  glifr_prep_input<<<8000, 256, 0, stream>>>(input, xbf);
  // zero state + hist ring (contiguous 1,572,864 B = 393,216 words)
  glifr_zero<<<1536, 256, 0, stream>>>((uint32_t*)(ws + 17301504), 393216);

  for (int c = 0; c < 50; ++c) {
    const int t0 = c * 20;
    glifr_chunk_kernel<<<16, 256, 131072, stream>>>(WcatT, xbf, hist,
                                                    volt_s, fire_s, asc_s,
                                                    thr, tkm, aamp, tar, tka, t0);
    glifr_out_kernel<<<40, 128, 0, stream>>>(hist, outwT, outb, out, t0);
  }
}
